// AttentionBlock_2937757630975
// MI455X (gfx1250) — matrix operand order repaired and checked by reading, not compiled, not run on hardware
//
#include <hip/hip_runtime.h>
#include <stdint.h>

// ---------------------------------------------------------------------------
// AttentionBlock for MI455X (gfx1250): bf16 WMMA flash attention.
//   x:[8,4096,256] f32.  q/k/v = x@W+b ; scores = qk^T * 1/16 ; softmax ;
//   o = attn@v ; out = x + o@Wp + bp
// ---------------------------------------------------------------------------

typedef __attribute__((ext_vector_type(16))) __bf16 v16bf;
typedef __attribute__((ext_vector_type(8)))  float  v8f;

union BF16x16  { v16bf v; uint4 q[2]; };
union BF16x16d { v16bf v; unsigned long long d[4]; };

__device__ __forceinline__ unsigned short f2bfu(float f) {
    union { float f; unsigned u; } x; x.f = f;
    unsigned r = x.u + 0x7fffu + ((x.u >> 16) & 1u);   // round-to-nearest-even
    return (unsigned short)(r >> 16);
}

#define NTOK 4096
#define CH   256

// ---------------------------------------------------------------------------
// elementwise f32 -> bf16
__global__ void cvt_bf16_kernel(const float* __restrict__ in,
                                unsigned short* __restrict__ out, int n) {
    int i = blockIdx.x * blockDim.x + threadIdx.x;
    if (i < n) out[i] = f2bfu(in[i]);
}

// transpose 256x256 weight [k][n] -> bf16 [n][k]
__global__ void transp_w_kernel(const float* __restrict__ w,
                                unsigned short* __restrict__ wt) {
    int n = blockIdx.x, k = threadIdx.x;
    wt[n * 256 + k] = f2bfu(w[k * 256 + n]);
}

// ---------------------------------------------------------------------------
// GEMM: Y[M x 256] = A[M x 256](bf16) @ W(256 x 256, given transposed [n][k] bf16)
// Then y = (acc + bias[n]) * scale ; store bf16, or f32 with residual add.
// One wave = 16 rows x 256 cols. 4 waves / block.
__global__ __launch_bounds__(128) void gemm256_kernel(
    const unsigned short* __restrict__ A,
    const unsigned short* __restrict__ Wt,
    const float* __restrict__ bias,
    float scale,
    unsigned short* __restrict__ outb,
    const float* __restrict__ resid,
    float* __restrict__ outf)
{
    const int tid  = threadIdx.x;
    const int wave = tid >> 5, lane = tid & 31;
    const int half = lane >> 4, ln = lane & 15;
    const long r0  = (long)(blockIdx.x * 4 + wave) * 16;

    // A tile (16 rows x 256) in A-layout chunks
    BF16x16 aa[8];
    const unsigned short* arow = A + (r0 + ln) * CH + 8 * half;
    #pragma unroll
    for (int c = 0; c < 8; ++c) {
        aa[c].q[0] = *(const uint4*)(arow + 32 * c);
        aa[c].q[1] = *(const uint4*)(arow + 32 * c + 16);
    }

    #pragma unroll
    for (int t = 0; t < 16; ++t) {
        v8f acc = {0.f,0.f,0.f,0.f,0.f,0.f,0.f,0.f};
        const unsigned short* wrow = Wt + (16 * t + ln) * CH + 8 * half;
        #pragma unroll
        for (int c = 0; c < 8; ++c) {
            BF16x16 wb;
            wb.q[0] = *(const uint4*)(wrow + 32 * c);
            wb.q[1] = *(const uint4*)(wrow + 32 * c + 16);
            acc = __builtin_amdgcn_wmma_f32_16x16x32_bf16(
                false, aa[c].v, false, wb.v, (short)0, acc, false, false);
        }
        const float bn = bias[16 * t + ln];
        #pragma unroll
        for (int j = 0; j < 8; ++j) {
            float v = (acc[j] + bn) * scale;
            long idx = (r0 + j + 8 * half) * CH + 16 * t + ln;
            if (outf) outf[idx] = resid[idx] + v;
            else      outb[idx] = f2bfu(v);
        }
    }
}

// ---------------------------------------------------------------------------
// Flash attention. Grid: (32 row-blocks, 8 batches), 256 threads (8 waves).
// Wave = 16 query rows; WG shares 32-key K/V tiles in LDS, software-pipelined:
// next tile is prefetched into registers while the current tile is computed.
// LDS map (56320 B):
//   [0     , 16896) K tile  : 32 keys x 264 bf16 (row pad -> conflict-free b128)
//   [16896 , 37376) V tile^T: 256 ch x 40 bf16, key granules XOR-swizzled by
//                   (chan>>5) to spread transpose-store banks
//   [37376 , 56320) per-wave scratch 2368 B: 16x36 f32 scores (P bf16 overlaid)
//                   + 16 f32 alpha
__global__ __launch_bounds__(256) void flash_attn_kernel(
    const unsigned short* __restrict__ Q,
    const unsigned short* __restrict__ K,
    const unsigned short* __restrict__ V,
    unsigned short* __restrict__ O)
{
    __shared__ __align__(16) unsigned char smem[56320];
    unsigned short* kt = (unsigned short*)smem;
    unsigned short* vt = (unsigned short*)(smem + 16896);

    const int tid  = threadIdx.x;
    const int wave = tid >> 5, lane = tid & 31;
    const int half = lane >> 4, ln = lane & 15;
    unsigned char* scr = smem + 37376 + wave * 2368;
    float* sbuf = (float*)scr;            // [16][36] f32 (rows 144B apart)
    float* abuf = (float*)(scr + 2304);   // [16] f32

    const int  b  = blockIdx.y;
    const int  m0 = blockIdx.x * 128 + wave * 16;
    const long base = (long)b * NTOK;

    // Q tile -> registers (A-layout)
    BF16x16 qa[8];
    {
        const unsigned short* qrow = Q + (base + m0 + ln) * CH + 8 * half;
        #pragma unroll
        for (int c = 0; c < 8; ++c) {
            qa[c].q[0] = *(const uint4*)(qrow + 32 * c);
            qa[c].q[1] = *(const uint4*)(qrow + 32 * c + 16);
        }
    }

    v8f oacc[16];
    #pragma unroll
    for (int t = 0; t < 16; ++t) oacc[t] = (v8f){0.f,0.f,0.f,0.f,0.f,0.f,0.f,0.f};
    float m_run = -3.0e38f, l_run = 0.0f;

    // tile loader mapping: thread stages 64B of one K row and one V row
    const int lkey = tid >> 3;            // 0..31  key this thread stages
    const int g    = tid & 7;             // channel-segment group
    const int lcs  = g * 32;              // channel segment base
    // XOR-swizzled key slot for the V^T store (granule = key>>2, swz = chan>>5 = g)
    const int kp   = (lkey & 3) | (((lkey >> 2) ^ g) << 2);

    uint4 kreg[4];
    union { uint4 q[4]; unsigned short h[32]; } vreg;
    {   // prefetch tile 0
        const uint4* ks = (const uint4*)(K + (base + lkey) * CH + lcs);
        const uint4* vs = (const uint4*)(V + (base + lkey) * CH + lcs);
        kreg[0] = ks[0]; kreg[1] = ks[1]; kreg[2] = ks[2]; kreg[3] = ks[3];
        vreg.q[0] = vs[0]; vreg.q[1] = vs[1]; vreg.q[2] = vs[2]; vreg.q[3] = vs[3];
    }

    for (int it = 0; it < NTOK / 32; ++it) {
        __syncthreads();                 // previous iteration's LDS reads done
        {   // commit staged K tile (row-major, padded rows)
            uint4* dst = (uint4*)(kt + lkey * 264 + lcs);
            dst[0] = kreg[0]; dst[1] = kreg[1]; dst[2] = kreg[2]; dst[3] = kreg[3];
        }
        {   // commit staged V tile, transposed + granule-swizzled
            #pragma unroll
            for (int j = 0; j < 32; ++j)
                vt[(lcs + j) * 40 + kp] = vreg.h[j];
        }
        if (it + 1 < NTOK / 32) {        // prefetch next tile (overlaps compute)
            const int n1 = (it + 1) * 32;
            const uint4* ks = (const uint4*)(K + (base + n1 + lkey) * CH + lcs);
            const uint4* vs = (const uint4*)(V + (base + n1 + lkey) * CH + lcs);
            kreg[0] = ks[0]; kreg[1] = ks[1]; kreg[2] = ks[2]; kreg[3] = ks[3];
            vreg.q[0] = vs[0]; vreg.q[1] = vs[1]; vreg.q[2] = vs[2]; vreg.q[3] = vs[3];
        }
        __syncthreads();                 // tiles visible to all waves

        // ---- scores: two 16x16 tiles over this 32-key window ----
        v8f s0 = {0.f,0.f,0.f,0.f,0.f,0.f,0.f,0.f};
        v8f s1 = {0.f,0.f,0.f,0.f,0.f,0.f,0.f,0.f};
        #pragma unroll
        for (int c = 0; c < 8; ++c) {
            BF16x16 kb;
            const unsigned short* kr0 = kt + ln * 264 + 32 * c + 8 * half;
            kb.q[0] = *(const uint4*)(kr0);
            kb.q[1] = *(const uint4*)(kr0 + 16);
            s0 = __builtin_amdgcn_wmma_f32_16x16x32_bf16(
                false, qa[c].v, false, kb.v, (short)0, s0, false, false);
            const unsigned short* kr1 = kt + (16 + ln) * 264 + 32 * c + 8 * half;
            kb.q[0] = *(const uint4*)(kr1);
            kb.q[1] = *(const uint4*)(kr1 + 16);
            s1 = __builtin_amdgcn_wmma_f32_16x16x32_bf16(
                false, qa[c].v, false, kb.v, (short)0, s1, false, false);
        }

        // ---- dump scores (C/D layout -> row-major LDS) ----
        #pragma unroll
        for (int j = 0; j < 8; ++j) {
            sbuf[(j + 8 * half) * 36 + ln]      = s0[j];
            sbuf[(j + 8 * half) * 36 + 16 + ln] = s1[j];
        }

        // ---- online softmax: lane owns row (ln); halves duplicate ----
        float4 rv[8];
        const float4* srow = (const float4*)(sbuf + ln * 36);
        #pragma unroll
        for (int i = 0; i < 8; ++i) rv[i] = srow[i];
        float tm = fmaxf(fmaxf(rv[0].x, rv[0].y), fmaxf(rv[0].z, rv[0].w));
        #pragma unroll
        for (int i = 1; i < 8; ++i)
            tm = fmaxf(tm, fmaxf(fmaxf(rv[i].x, rv[i].y), fmaxf(rv[i].z, rv[i].w)));
        const float m_new = fmaxf(m_run, tm);
        const float alpha = __expf(m_run - m_new);
        float rs = 0.f;
        unsigned* prow = (unsigned*)((unsigned char*)sbuf + ln * 144); // P overlays own row
        #pragma unroll
        for (int i = 0; i < 8; ++i) {
            float p0 = __expf(rv[i].x - m_new);
            float p1 = __expf(rv[i].y - m_new);
            float p2 = __expf(rv[i].z - m_new);
            float p3 = __expf(rv[i].w - m_new);
            rs += (p0 + p1) + (p2 + p3);
            prow[2 * i]     = (unsigned)f2bfu(p0) | ((unsigned)f2bfu(p1) << 16);
            prow[2 * i + 1] = (unsigned)f2bfu(p2) | ((unsigned)f2bfu(p3) << 16);
        }
        l_run = l_run * alpha + rs;
        m_run = m_new;
        abuf[ln] = alpha;

        // ---- rescale O accumulators by per-row alpha ----
        float4 a0 = *(const float4*)(abuf + 8 * half);
        float4 a1 = *(const float4*)(abuf + 8 * half + 4);
        v8f am = {a0.x, a0.y, a0.z, a0.w, a1.x, a1.y, a1.z, a1.w};
        #pragma unroll
        for (int t = 0; t < 16; ++t) oacc[t] *= am;

        // ---- P (A-layout) from LDS, then O += P @ V ----
        BF16x16 pa;
        const unsigned char* pb = (const unsigned char*)sbuf + ln * 144 + 32 * half;
        pa.q[0] = *(const uint4*)(pb);
        pa.q[1] = *(const uint4*)(pb + 16);
        #pragma unroll
        for (int t = 0; t < 16; ++t) {
            const int c = 16 * t + ln;
            const unsigned short* vrow = vt + c * 40;
            const int swz = (c >> 5) & 7;
            BF16x16d vb;     // un-swizzle: granule gi lives at slot gi ^ swz
            vb.d[0] = *(const unsigned long long*)(vrow + (((4 * half + 0) ^ swz) << 2));
            vb.d[1] = *(const unsigned long long*)(vrow + (((4 * half + 1) ^ swz) << 2));
            vb.d[2] = *(const unsigned long long*)(vrow + (((4 * half + 2) ^ swz) << 2));
            vb.d[3] = *(const unsigned long long*)(vrow + (((4 * half + 3) ^ swz) << 2));
            oacc[t] = __builtin_amdgcn_wmma_f32_16x16x32_bf16(
                false, pa.v, false, vb.v, (short)0, oacc[t], false, false);
        }
    }

    // ---- finalize: O /= l ; store bf16 ----
    abuf[ln] = 1.0f / l_run;
    float4 a0 = *(const float4*)(abuf + 8 * half);
    float4 a1 = *(const float4*)(abuf + 8 * half + 4);
    const float inv[8] = {a0.x, a0.y, a0.z, a0.w, a1.x, a1.y, a1.z, a1.w};
    #pragma unroll
    for (int t = 0; t < 16; ++t) {
        #pragma unroll
        for (int j = 0; j < 8; ++j) {
            float v = oacc[t][j] * inv[j];
            O[(base + m0 + j + 8 * half) * CH + 16 * t + ln] = f2bfu(v);
        }
    }
}

// ---------------------------------------------------------------------------
extern "C" void kernel_launch(void* const* d_in, const int* in_sizes, int n_in,
                              void* d_out, int out_size, void* d_ws, size_t ws_size,
                              hipStream_t stream) {
    const float* x  = (const float*)d_in[0];
    const float* wq = (const float*)d_in[1]; const float* bq = (const float*)d_in[2];
    const float* wk = (const float*)d_in[3]; const float* bk = (const float*)d_in[4];
    const float* wv = (const float*)d_in[5]; const float* bv = (const float*)d_in[6];
    const float* wp = (const float*)d_in[7]; const float* bp = (const float*)d_in[8];
    float* out = (float*)d_out;

    const long NE = 8L * NTOK * CH;              // 8388608 elements
    unsigned short* xbf = (unsigned short*)d_ws; // bf16 buffers in workspace
    unsigned short* qbf = xbf + NE;
    unsigned short* kbf = qbf + NE;
    unsigned short* vbf = kbf + NE;
    unsigned short* obf = vbf + NE;
    unsigned short* wtq = obf + NE;              // 65536 each (transposed bf16)
    unsigned short* wtk = wtq + 65536;
    unsigned short* wtv = wtk + 65536;
    unsigned short* wtp = wtv + 65536;

    cvt_bf16_kernel<<<(int)((NE + 255) / 256), 256, 0, stream>>>(x, xbf, (int)NE);
    transp_w_kernel<<<256, 256, 0, stream>>>(wq, wtq);
    transp_w_kernel<<<256, 256, 0, stream>>>(wk, wtk);
    transp_w_kernel<<<256, 256, 0, stream>>>(wv, wtv);
    transp_w_kernel<<<256, 256, 0, stream>>>(wp, wtp);

    // Q gets the 1/sqrt(256) = 1/16 score scale folded in (applied after bias).
    gemm256_kernel<<<512, 128, 0, stream>>>(xbf, wtq, bq, 0.0625f, qbf, nullptr, nullptr);
    gemm256_kernel<<<512, 128, 0, stream>>>(xbf, wtk, bk, 1.0f,    kbf, nullptr, nullptr);
    gemm256_kernel<<<512, 128, 0, stream>>>(xbf, wtv, bv, 1.0f,    vbf, nullptr, nullptr);

    flash_attn_kernel<<<dim3(32, 8), 256, 0, stream>>>(qbf, kbf, vbf, obf);

    // out = x + O @ Wp + bp
    gemm256_kernel<<<512, 128, 0, stream>>>(obf, wtp, bp, 1.0f, nullptr, x, out);
}
